// GCNEncoder_39152921870553
// MI455X (gfx1250) — matrix operand order repaired and checked by reading, not compiled, not run on hardware
//
#include <hip/hip_runtime.h>

typedef __attribute__((ext_vector_type(16))) _Float16 v16h;
typedef __attribute__((ext_vector_type(8)))  _Float16 v8h;
typedef __attribute__((ext_vector_type(8)))  float    v8f;

#define BB   8
#define NPTS 2048
#define KNN  10

__device__ __forceinline__ float leaky(float x) { return x >= 0.f ? x : 0.2f * x; }

// ---------------- weight prep: f32 (Otrue x Itrue) -> f16 (Opad x Ipad), pad-remapped ----
__global__ void k_prepw(const float* __restrict__ src, _Float16* __restrict__ dst,
                        int Otrue, int Opad, int Itrue, int Ipad,
                        int splitTrue, int splitOff) {
  long tid = (long)blockIdx.x * blockDim.x + threadIdx.x;
  if (tid >= (long)Opad * Ipad) return;
  int ip = (int)(tid % Ipad);
  int o  = (int)(tid / Ipad);
  float w = 0.f;
  if (o < Otrue) {
    int i = -1;
    if (ip < splitTrue) i = ip;
    else if (ip >= splitOff) {
      int ii = ip - splitOff + splitTrue;
      if (ii < Itrue) i = ii;
    }
    if (i >= 0) w = src[(size_t)o * Itrue + i];
  }
  dst[tid] = (_Float16)w;
}

// ---------------- transpose input (B,64,N) f32 -> (B*N, 64) f32 + f16 --------------------
__global__ void k_transpose(const float* __restrict__ x, float* __restrict__ xt32,
                            _Float16* __restrict__ xt16) {
  long tid = (long)blockIdx.x * blockDim.x + threadIdx.x;   // (b, c, n), n fastest
  if (tid >= (long)BB * 64 * NPTS) return;
  int n = (int)(tid % NPTS);
  long t = tid / NPTS;
  int c = (int)(t % 64);
  int b = (int)(t / 64);
  float v = x[tid];
  size_t o = ((size_t)b * NPTS + n) * 64 + c;
  xt32[o] = v;
  xt16[o] = (_Float16)v;
}

// ---------------- per-point squared norms -----------------------------------------------
__global__ void k_sqnorm(const float* __restrict__ pts, float* __restrict__ sq,
                         int rows, int Cs) {
  int r = blockIdx.x * blockDim.x + threadIdx.x;
  if (r >= rows) return;
  const float* p = pts + (size_t)r * Cs;
  float s = 0.f;
  for (int c = 0; c < Cs; ++c) s += p[c] * p[c];
  sq[r] = s;
}

// ---------------- G[b][n][m] = -2 * dot(pts_n, pts_m), fp32 LDS-tiled --------------------
__global__ __launch_bounds__(256) void k_dist(const float* __restrict__ pts,
                                              float* __restrict__ G, int Cs) {
  __shared__ float As[16][33];
  __shared__ float Bs[16][33];
  int b  = blockIdx.z;
  int n0 = blockIdx.y * 16, m0 = blockIdx.x * 16;
  int tx = threadIdx.x & 15, ty = threadIdx.x >> 4;
  const float* pb = pts + (size_t)b * NPTS * Cs;
  float acc = 0.f;
  for (int c0 = 0; c0 < Cs; c0 += 32) {
    As[ty][tx]      = pb[(size_t)(n0 + ty) * Cs + c0 + tx];
    As[ty][tx + 16] = pb[(size_t)(n0 + ty) * Cs + c0 + tx + 16];
    Bs[ty][tx]      = pb[(size_t)(m0 + ty) * Cs + c0 + tx];
    Bs[ty][tx + 16] = pb[(size_t)(m0 + ty) * Cs + c0 + tx + 16];
    __syncthreads();
#pragma unroll
    for (int cc = 0; cc < 32; ++cc) acc += As[ty][cc] * Bs[tx][cc];
    __syncthreads();
  }
  G[(size_t)b * NPTS * NPTS + (size_t)(n0 + ty) * NPTS + m0 + tx] = -2.f * acc;
}

// ---------------- top-10 nearest (excluding self); coalesced column scan -----------------
__global__ void k_topk(const float* __restrict__ G, const float* __restrict__ sq,
                       int* __restrict__ idx) {
  int b = blockIdx.y;
  int n = blockIdx.x * blockDim.x + threadIdx.x;
  if (n >= NPTS) return;
  const float* g   = G  + (size_t)b * NPTS * NPTS;
  const float* sqb = sq + (size_t)b * NPTS;
  float bd[KNN];
  int   bi[KNN];
#pragma unroll
  for (int j = 0; j < KNN; ++j) { bd[j] = 3.0e38f; bi[j] = 0; }
  for (int m = 0; m < NPTS; ++m) {
    if (m == n) continue;
    float d = g[(size_t)m * NPTS + n] + sqb[m];   // symmetric dot + sq[m]; sq[n] const
    if (d < bd[KNN - 1]) {
      int j = KNN - 1;
      while (j > 0 && bd[j - 1] > d) { bd[j] = bd[j - 1]; bi[j] = bi[j - 1]; --j; }
      bd[j] = d; bi[j] = m;
    }
  }
  int* out = idx + ((size_t)b * NPTS + n) * KNN;
#pragma unroll
  for (int j = 0; j < KNN; ++j) out[j] = bi[j];
}

// ---------------- edge features: (B*N*K, 2*Cp) = [nbr - cen | cen], f16 ------------------
__global__ void k_feat(const _Float16* __restrict__ pts, const int* __restrict__ idx,
                       _Float16* __restrict__ feat, int Cp) {
  long tid = (long)blockIdx.x * blockDim.x + threadIdx.x;
  int  g8  = Cp >> 3;
  long total = (long)BB * NPTS * KNN * g8;
  if (tid >= total) return;
  int  c8  = (int)(tid % g8);
  long col = tid / g8;                       // (b*N + n)*KNN + k
  long bn  = col / KNN;
  int  k   = (int)(col % KNN);
  int  b   = (int)(bn / NPTS);
  int  m   = idx[bn * KNN + k];
  v8h cen = ((const v8h*)(pts + (size_t)bn * Cp))[c8];
  v8h nb  = ((const v8h*)(pts + ((size_t)b * NPTS + m) * Cp))[c8];
  v8h df;
#pragma unroll
  for (int j = 0; j < 8; ++j) df[j] = (_Float16)(nb[j] - cen[j]);
  v8h* out = (v8h*)(feat + (size_t)col * (2 * Cp));
  out[c8]      = df;
  out[g8 + c8] = cen;
}

// ---------------- WMMA conv: Y(CP,Opad) = X(CP,Ipad) x W(Opad,Ipad)^T, + BN stats --------
// Each wave owns a 16-output-channel stripe x 8 consecutive 16-column tiles.
// K outermost: one A (weight) fragment load feeds 8 back-to-back WMMAs.
// Grid covers columns exactly (CP is a multiple of 1024), so EXEC stays full.
__global__ __launch_bounds__(256) void k_conv_wmma(
    const _Float16* __restrict__ X, const _Float16* __restrict__ W,
    _Float16* __restrict__ Y, float* __restrict__ stats,
    int CP, int Ipad, int Opad) {
  const int lane = threadIdx.x & 31;
  const int wave = threadIdx.x >> 5;
  const int half = lane >> 4;
  const int l15  = lane & 15;
  const int o0   = blockIdx.x * 16;
  const long ptile0 = ((long)blockIdx.y * 8 + wave) * 8;

  const _Float16* wrow  = W + (size_t)(o0 + l15) * Ipad;           // A row M = l15
  const _Float16* xbase = X + (size_t)((ptile0 << 4) + l15) * Ipad; // B col N = l15
  const size_t tileStride = (size_t)16 * Ipad;                      // next 16-col tile

  v8f c[8];
#pragma unroll
  for (int t = 0; t < 8; ++t) c[t] = (v8f){};

  for (int k0 = 0; k0 < Ipad; k0 += 32) {
    // A fragment: lane holds row M=l15; elem e -> K = half*16 + e (contiguous)
    v8h alo = *(const v8h*)(wrow + k0 + half * 16);
    v8h ahi = *(const v8h*)(wrow + k0 + half * 16 + 8);
    v16h a;
#pragma unroll
    for (int j = 0; j < 8; ++j) { a[j] = alo[j]; a[j + 8] = ahi[j]; }
    // B fragment: lane holds col N=l15; elem e -> K = half*16 + e (contiguous 32B)
#pragma unroll
    for (int t = 0; t < 8; ++t) {
      v16h bfrag = *(const v16h*)(xbase + (size_t)t * tileStride + k0 + half * 16);
      c[t] = __builtin_amdgcn_wmma_f32_16x16x32_f16(false, a, false, bfrag,
                                                    (short)0, c[t], false, false);
    }
  }

  // Epilogue: store f16 result, accumulate BN sums per channel group
  float sAcc[8], qAcc[8];
#pragma unroll
  for (int r = 0; r < 8; ++r) { sAcc[r] = 0.f; qAcc[r] = 0.f; }
#pragma unroll
  for (int t = 0; t < 8; ++t) {
    long P = ((ptile0 + t) << 4) + l15;       // D: lane covers column P
    v8h outv;
#pragma unroll
    for (int r = 0; r < 8; ++r) {             // channels o = o0 + half*8 + r
      float v = c[t][r];
      sAcc[r] += v; qAcc[r] += v * v;
      outv[r] = (_Float16)v;
    }
    *(v8h*)(Y + (size_t)P * Opad + o0 + half * 8) = outv;
  }
  // reduce BN stats across the 16 lanes sharing a channel group, then atomics
#pragma unroll
  for (int r = 0; r < 8; ++r) {
    float s = sAcc[r], q = qAcc[r];
    for (int m = 1; m < 16; m <<= 1) { s += __shfl_xor(s, m); q += __shfl_xor(q, m); }
    if (l15 == 0) {
      int o = o0 + half * 8 + r;
      atomicAdd(&stats[o], s);
      atomicAdd(&stats[Opad + o], q);
    }
  }
  (void)CP;
}

// ---------------- BN param finalize: prm[o]=scale, prm[Opad+o]=shift ---------------------
__global__ void k_bnparams(const float* __restrict__ stats, const float* __restrict__ g,
                           const float* __restrict__ be, float* __restrict__ prm,
                           int Otrue, int Opad, float invCnt) {
  int o = blockIdx.x * blockDim.x + threadIdx.x;
  if (o >= Opad) return;
  if (o < Otrue) {
    float m = stats[o] * invCnt;
    float v = stats[Opad + o] * invCnt - m * m;
    float s = g[o] * rsqrtf(v + 1e-5f);
    prm[o] = s; prm[Opad + o] = be[o] - m * s;
  } else {
    prm[o] = 0.f; prm[Opad + o] = 0.f;
  }
}

__global__ void k_zero(float* p, int n) {
  int i = blockIdx.x * blockDim.x + threadIdx.x;
  if (i < n) p[i] = 0.f;
}

// ---------------- in-place BN + leaky, f16 (cols, Opad) ----------------------------------
__global__ void k_act(_Float16* __restrict__ Y, const float* __restrict__ prm,
                      long cols, int Opad) {
  long tid = (long)blockIdx.x * blockDim.x + threadIdx.x;
  int g8 = Opad >> 3;
  if (tid >= cols * g8) return;
  int  c8  = (int)(tid % g8);
  long col = tid / g8;
  v8h* p = (v8h*)(Y + (size_t)col * Opad) + c8;
  v8h v = *p;
#pragma unroll
  for (int j = 0; j < 8; ++j) {
    int o = c8 * 8 + j;
    v[j] = (_Float16)leaky((float)v[j] * prm[o] + prm[Opad + o]);
  }
  *p = v;
}

// ---------------- BN + leaky + max over K -> f16 (+optional f32) (B*N, Opad) -------------
__global__ void k_act_max(const _Float16* __restrict__ Y, const float* __restrict__ prm,
                          _Float16* __restrict__ out16, float* __restrict__ out32,
                          long ncols, int Opad) {
  long tid = (long)blockIdx.x * blockDim.x + threadIdx.x;
  int g8 = Opad >> 3;
  if (tid >= ncols * g8) return;
  int  c8 = (int)(tid % g8);
  long bn = tid / g8;
  float mx[8];
#pragma unroll
  for (int j = 0; j < 8; ++j) mx[j] = -3.0e38f;
  for (int k = 0; k < KNN; ++k) {
    v8h v = *((const v8h*)(Y + ((size_t)bn * KNN + k) * Opad) + c8);
#pragma unroll
    for (int j = 0; j < 8; ++j) {
      int o = c8 * 8 + j;
      float x = leaky((float)v[j] * prm[o] + prm[Opad + o]);
      mx[j] = fmaxf(mx[j], x);
    }
  }
  v8h o16;
#pragma unroll
  for (int j = 0; j < 8; ++j) o16[j] = (_Float16)mx[j];
  *((v8h*)(out16 + (size_t)bn * Opad) + c8) = o16;
  if (out32) {
#pragma unroll
    for (int j = 0; j < 8; ++j) out32[(size_t)bn * Opad + c8 * 8 + j] = mx[j];
  }
}

// ---------------- concat xc(B*N,704) = [x1(192) | x2(512)] -------------------------------
__global__ void k_concat(const _Float16* __restrict__ x1, const _Float16* __restrict__ x2,
                         _Float16* __restrict__ xc, long rows) {
  long tid = (long)blockIdx.x * blockDim.x + threadIdx.x;
  if (tid >= rows * 88) return;
  int  c8  = (int)(tid % 88);
  long row = tid / 88;
  v8h v = (c8 < 24) ? ((const v8h*)(x1 + (size_t)row * 192))[c8]
                    : ((const v8h*)(x2 + (size_t)row * 512))[c8 - 24];
  ((v8h*)(xc + (size_t)row * 704))[c8] = v;
}

// ---------------- final BN + leaky -> d_out (B, 512, N) f32 ------------------------------
__global__ void k_final(const _Float16* __restrict__ Y, const float* __restrict__ prm,
                        float* __restrict__ out) {
  long tid = (long)blockIdx.x * blockDim.x + threadIdx.x;  // (b, o, n), n fastest
  if (tid >= (long)BB * 512 * NPTS) return;
  int  n = (int)(tid % NPTS);
  long t = tid / NPTS;
  int  o = (int)(t % 512);
  int  b = (int)(t / 512);
  float x = (float)Y[((size_t)b * NPTS + n) * 512 + o];
  out[tid] = leaky(x * prm[o] + prm[512 + o]);
}

// ========================================================================================
extern "C" void kernel_launch(void* const* d_in, const int* in_sizes, int n_in,
                              void* d_out, int out_size, void* d_ws, size_t ws_size,
                              hipStream_t stream) {
  (void)in_sizes; (void)n_in; (void)out_size; (void)ws_size;
  const long BN  = (long)BB * NPTS;      // 16384
  const long BNK = BN * KNN;             // 163840

  const float* x    = (const float*)d_in[0];
  const float* w11  = (const float*)d_in[1];
  const float* g11  = (const float*)d_in[3];
  const float* be11 = (const float*)d_in[4];
  const float* w12  = (const float*)d_in[5];
  const float* g12  = (const float*)d_in[7];
  const float* be12 = (const float*)d_in[8];
  const float* w21  = (const float*)d_in[9];
  const float* g21  = (const float*)d_in[11];
  const float* be21 = (const float*)d_in[12];
  const float* w22  = (const float*)d_in[13];
  const float* g22  = (const float*)d_in[15];
  const float* be22 = (const float*)d_in[16];
  const float* w31  = (const float*)d_in[17];
  const float* g31  = (const float*)d_in[19];
  const float* be31 = (const float*)d_in[20];
  const float* w32  = (const float*)d_in[21];
  const float* g32  = (const float*)d_in[23];
  const float* be32 = (const float*)d_in[24];

  char* base = (char*)d_ws;
  size_t off = 0;
  auto carve = [&](size_t bytes) -> char* {
    off = (off + 255) & ~(size_t)255;
    char* p = base + off; off += bytes; return p;
  };

  _Float16* W11 = (_Float16*)carve((size_t)160 * 128 * 2);
  _Float16* W12 = (_Float16*)carve((size_t)192 * 160 * 2);
  _Float16* W21 = (_Float16*)carve((size_t)448 * 384 * 2);
  _Float16* W22 = (_Float16*)carve((size_t)512 * 448 * 2);
  _Float16* W31 = (_Float16*)carve((size_t)608 * 704 * 2);
  _Float16* W32 = (_Float16*)carve((size_t)512 * 608 * 2);
  float*    xt32  = (float*)carve((size_t)BN * 64 * 4);
  _Float16* xt16  = (_Float16*)carve((size_t)BN * 64 * 2);
  float*    x1_32 = (float*)carve((size_t)BN * 192 * 4);
  _Float16* x1_16 = (_Float16*)carve((size_t)BN * 192 * 2);
  _Float16* x2_16 = (_Float16*)carve((size_t)BN * 512 * 2);
  _Float16* xc    = (_Float16*)carve((size_t)BN * 704 * 2);
  float*    sqb   = (float*)carve((size_t)BN * 4);
  int*      idx   = (int*)carve((size_t)BN * KNN * 4);
  float*    stats = (float*)carve(1280 * 4);
  float*    prm   = (float*)carve(1280 * 4);
  _Float16* feat  = (_Float16*)carve((size_t)BNK * 384 * 2);
  char* big = carve((size_t)BNK * 448 * 2 + (size_t)BNK * 512 * 2);   // ~314 MB slab
  _Float16* ya = (_Float16*)big;
  _Float16* yb = (_Float16*)(big + (size_t)BNK * 448 * 2);
  float*    G  = (float*)big;            // 134 MB distance matrix aliases ya/yb (disjoint lifetime)

  auto grid1 = [](long total) { return dim3((unsigned)((total + 255) / 256)); };
  auto conv = [&](const _Float16* X, const _Float16* W, _Float16* Y,
                  long CP, int Ipad, int Opad) {
    dim3 g(Opad / 16, (unsigned)(CP / 1024), 1);   // CP is a multiple of 1024
    k_conv_wmma<<<g, 256, 0, stream>>>(X, W, Y, stats, (int)CP, Ipad, Opad);
  };
  auto zero = [&]() { k_zero<<<grid1(1280), 256, 0, stream>>>(stats, 1280); };
  auto bnp = [&](const float* gg, const float* bb, int Otrue, int Opad, long cnt) {
    k_bnparams<<<grid1(Opad), 256, 0, stream>>>(stats, gg, bb, prm, Otrue, Opad,
                                                1.0f / (float)cnt);
  };

  // weights -> f16, padded/remapped (split at 181->192 for edge-feature concat layouts)
  k_prepw<<<grid1(160L * 128), 256, 0, stream>>>(w11, W11, 152, 160, 128, 128, 128, 128);
  k_prepw<<<grid1(192L * 160), 256, 0, stream>>>(w12, W12, 181, 192, 152, 160, 152, 160);
  k_prepw<<<grid1(448L * 384), 256, 0, stream>>>(w21, W21, 430, 448, 362, 384, 181, 192);
  k_prepw<<<grid1(512L * 448), 256, 0, stream>>>(w22, W22, 512, 512, 430, 448, 430, 448);
  k_prepw<<<grid1(608L * 704), 256, 0, stream>>>(w31, W31, 595, 608, 693, 704, 181, 192);
  k_prepw<<<grid1(512L * 608), 256, 0, stream>>>(w32, W32, 512, 512, 595, 608, 595, 608);

  // ---- block 1 -------------------------------------------------------------------------
  k_transpose<<<grid1((long)BB * 64 * NPTS), 256, 0, stream>>>(x, xt32, xt16);
  k_sqnorm<<<grid1(BN), 256, 0, stream>>>(xt32, sqb, (int)BN, 64);
  k_dist<<<dim3(NPTS / 16, NPTS / 16, BB), 256, 0, stream>>>(xt32, G, 64);
  k_topk<<<dim3(NPTS / 256, BB), 256, 0, stream>>>(G, sqb, idx);
  k_feat<<<grid1(BNK * 8), 256, 0, stream>>>(xt16, idx, feat, 64);

  zero(); conv(feat, W11, ya, BNK, 128, 160);
  bnp(g11, be11, 152, 160, BNK);
  k_act<<<grid1(BNK * 20), 256, 0, stream>>>(ya, prm, BNK, 160);

  zero(); conv(ya, W12, yb, BNK, 160, 192);
  bnp(g12, be12, 181, 192, BNK);
  k_act_max<<<grid1(BN * 24), 256, 0, stream>>>(yb, prm, x1_16, x1_32, BN, 192);

  // ---- block 2 -------------------------------------------------------------------------
  k_sqnorm<<<grid1(BN), 256, 0, stream>>>(x1_32, sqb, (int)BN, 192);
  k_dist<<<dim3(NPTS / 16, NPTS / 16, BB), 256, 0, stream>>>(x1_32, G, 192);
  k_topk<<<dim3(NPTS / 256, BB), 256, 0, stream>>>(G, sqb, idx);
  k_feat<<<grid1(BNK * 24), 256, 0, stream>>>(x1_16, idx, feat, 192);

  zero(); conv(feat, W21, ya, BNK, 384, 448);
  bnp(g21, be21, 430, 448, BNK);
  k_act<<<grid1(BNK * 56), 256, 0, stream>>>(ya, prm, BNK, 448);

  zero(); conv(ya, W22, yb, BNK, 448, 512);
  bnp(g22, be22, 512, 512, BNK);
  k_act_max<<<grid1(BN * 64), 256, 0, stream>>>(yb, prm, x2_16, nullptr, BN, 512);

  // ---- block 3 -------------------------------------------------------------------------
  k_concat<<<grid1(BN * 88), 256, 0, stream>>>(x1_16, x2_16, xc, BN);

  zero(); conv(xc, W31, ya, BN, 704, 608);
  bnp(g31, be31, 595, 608, BN);
  k_act<<<grid1(BN * 76), 256, 0, stream>>>(ya, prm, BN, 608);

  zero(); conv(ya, W32, yb, BN, 608, 512);
  bnp(g32, be32, 512, 512, BN);
  k_final<<<grid1((long)BB * 512 * NPTS), 256, 0, stream>>>(yb, prm, (float*)d_out);
}
